// TransformerBlock_55748675502751
// MI455X (gfx1250) — compile-verified, matrix operand order checked by reading, not run on hardware
//
#include <hip/hip_runtime.h>
#include <hip/hip_bf16.h>
#include <stdint.h>

// ---------------------------------------------------------------------------
// Transformer block for MI455X (gfx1250, wave32, WMMA + async-to-LDS).
// B=4, C=2048, D=1024, H=16, dh=64, HIDDEN=4096. fp32 in/out, f16 WMMA inside.
// ---------------------------------------------------------------------------

typedef __attribute__((ext_vector_type(16))) _Float16 v16h;
typedef __attribute__((ext_vector_type(8)))  _Float16 v8h;
typedef __attribute__((ext_vector_type(8)))  float    v8f;

#define WMMA_F16(a, b, c) \
  __builtin_amdgcn_wmma_f32_16x16x32_f16(false, (a), false, (b), (short)0, (c), false, false)

// Async DMA of 16 bytes per lane: global -> LDS, tracked by ASYNCcnt.
__device__ __forceinline__ void async_cp16(uint32_t lds_off, const void* gptr) {
  uint64_t g = (uint64_t)(uintptr_t)gptr;
  asm volatile("global_load_async_to_lds_b128 %0, %1, off" :: "v"(lds_off), "v"(g) : "memory");
}
__device__ __forceinline__ void wait_async0() {
  asm volatile("s_wait_asynccnt 0x0" ::: "memory");
}

static constexpr int CB  = 4;
static constexpr int CC  = 2048;
static constexpr int CD  = 1024;
static constexpr int CH  = 16;
static constexpr int CDH = 64;
static constexpr int CHID = 4096;
static constexpr int CM  = CB * CC; // 8192 rows

// Load one 16x32 f16 fragment (A layout; also used for B with "row" = N index,
// data stored N-major with K contiguous). Per ISA 7.12.2:
//   lane l: row = l&15 ; elem i: k = 16*(i/8) + 8*(l>>4) + (i%8)
__device__ __forceinline__ v16h frag_ld(const _Float16* __restrict__ tile, int ld) {
  const int lane = threadIdx.x & 31;
  const _Float16* p = tile + (size_t)(lane & 15) * ld + ((lane >> 4) << 3);
  v8h lo = *(const v8h*)p;
  v8h hi = *(const v8h*)(p + 16);
  v16h f;
#pragma unroll
  for (int i = 0; i < 8; ++i) { f[i] = lo[i]; f[i + 8] = hi[i]; }
  return f;
}

// ---------------------------------------------------------------------------
// Weight convert + transpose: W (K x N, f32 row-major) -> Wt (N x K, f16)
// ---------------------------------------------------------------------------
__global__ __launch_bounds__(256) void wt_f16_kernel(const float* __restrict__ W,
                                                     _Float16* __restrict__ Wt,
                                                     int K, int N) {
  size_t idx = (size_t)blockIdx.x * blockDim.x + threadIdx.x;
  size_t total = (size_t)K * N;
  if (idx < total) {
    int n = (int)(idx / K);
    int k = (int)(idx % K);
    Wt[idx] = (_Float16)W[(size_t)k * N + n];
  }
}

// ---------------------------------------------------------------------------
// LayerNorm: one 256-thread block per row of 1024. f32 in, f16 out.
// ---------------------------------------------------------------------------
__global__ __launch_bounds__(256) void layernorm_kernel(const float* __restrict__ x,
                                                        const float* __restrict__ gamma,
                                                        const float* __restrict__ beta,
                                                        _Float16* __restrict__ y) {
  __shared__ float rs[8], rss[8];
  const int row = blockIdx.x;
  const float* xr = x + (size_t)row * CD;
  float s = 0.f, ss = 0.f;
  for (int i = threadIdx.x; i < CD; i += 256) { float v = xr[i]; s += v; ss += v * v; }
#pragma unroll
  for (int off = 1; off < 32; off <<= 1) {
    s  += __shfl_xor(s,  off, 32);
    ss += __shfl_xor(ss, off, 32);
  }
  const int lane = threadIdx.x & 31, wv = threadIdx.x >> 5;
  if (lane == 0) { rs[wv] = s; rss[wv] = ss; }
  __syncthreads();
  s = 0.f; ss = 0.f;
#pragma unroll
  for (int i = 0; i < 8; ++i) { s += rs[i]; ss += rss[i]; }
  const float mean = s * (1.f / CD);
  const float var  = ss * (1.f / CD) - mean * mean;
  const float inv  = rsqrtf(var + 1e-5f);
  _Float16* yr = y + (size_t)row * CD;
  for (int i = threadIdx.x; i < CD; i += 256)
    yr[i] = (_Float16)((xr[i] - mean) * inv * gamma[i] + beta[i]);
}

// ---------------------------------------------------------------------------
// WMMA GEMM with double-buffered async-to-LDS staging.
// C[M,N] = epilogue(A[M,K] @ B[K,N] + bias (+resid))
// A: f16 row-major (ld=K). Bt: f16 N-major (ld=K). bias: f32[N].
// OUTMODE: 0 = f32 out, 1 = f16 out, 2 = QKV scatter.
// Block: 256 thr = 8 waves. Block tile 128(M) x 64(N), K-step 64.
// LDS: 2 x (128x64 A) + 2 x (64x64 B) f16 = 48 KB.
// ---------------------------------------------------------------------------
template <int OUTMODE, bool GELU_ACT, bool RESID>
__global__ __launch_bounds__(256) void gemm_wmma_kernel(
    const _Float16* __restrict__ A, const _Float16* __restrict__ Bt,
    const float* __restrict__ bias, const float* __restrict__ resid,
    float* __restrict__ outF, _Float16* __restrict__ outH,
    _Float16* __restrict__ qo, _Float16* __restrict__ ko, _Float16* __restrict__ vto,
    int M, int N, int K) {
  __shared__ __align__(16) _Float16 Ab[2][128 * 64];
  __shared__ __align__(16) _Float16 Bb[2][64 * 64];

  const int tid  = threadIdx.x;
  const int lane = tid & 31;
  const int wave = tid >> 5;
  const int mw = wave >> 1;            // 0..3 -> 32-row slice
  const int nw = wave & 1;             // 0..1 -> 32-col slice
  const int m0 = blockIdx.y * 128;
  const int n0 = blockIdx.x * 64;
  const int hl = lane >> 4, ln = lane & 15;

  // per-lane copy coordinates: each lane moves 16B chunks of a 64-half row
  const int crow = tid >> 3;           // 0..31 (row within a 32-row slab)
  const int ccol = (tid & 7) << 3;     // half-offset 0,8,...,56

  const uint32_t AbL0 = (uint32_t)(uintptr_t)&Ab[0][0];
  const uint32_t AbL1 = (uint32_t)(uintptr_t)&Ab[1][0];
  const uint32_t BbL0 = (uint32_t)(uintptr_t)&Bb[0][0];
  const uint32_t BbL1 = (uint32_t)(uintptr_t)&Bb[1][0];

  v8f acc00 = {}, acc01 = {}, acc10 = {}, acc11 = {};

  auto issue_tiles = [&](int buf, int k0) {
    const uint32_t aL = buf ? AbL1 : AbL0;
    const uint32_t bL = buf ? BbL1 : BbL0;
#pragma unroll
    for (int i = 0; i < 4; ++i) {      // A: 128 rows x 64 halves
      const int row = crow + 32 * i;
      async_cp16(aL + (uint32_t)(row * 64 + ccol) * 2,
                 A + (size_t)(m0 + row) * K + k0 + ccol);
    }
#pragma unroll
    for (int i = 0; i < 2; ++i) {      // B: 64 rows x 64 halves
      const int row = crow + 32 * i;
      async_cp16(bL + (uint32_t)(row * 64 + ccol) * 2,
                 Bt + (size_t)(n0 + row) * K + k0 + ccol);
    }
  };

  issue_tiles(0, 0);
  const int nk = K >> 6;
  for (int kt = 0; kt < nk; ++kt) {
    const int buf = kt & 1;
    wait_async0();                     // own async copies into buf complete
    __syncthreads();                   // whole block's tile ready
    if (kt + 1 < nk) issue_tiles(buf ^ 1, (kt + 1) << 6);

    const _Float16* As = &Ab[buf][mw * 32 * 64];
    const _Float16* Bs = &Bb[buf][nw * 32 * 64];
#pragma unroll
    for (int kk = 0; kk < 64; kk += 32) {
      v16h a0 = frag_ld(As + kk, 64);
      v16h a1 = frag_ld(As + 16 * 64 + kk, 64);
      v16h b0 = frag_ld(Bs + kk, 64);
      v16h b1 = frag_ld(Bs + 16 * 64 + kk, 64);
      acc00 = WMMA_F16(a0, b0, acc00);
      acc01 = WMMA_F16(a0, b1, acc01);
      acc10 = WMMA_F16(a1, b0, acc10);
      acc11 = WMMA_F16(a1, b1, acc11);
    }
    __syncthreads();                   // reads of buf done before it is re-filled
  }

  const int wm0 = m0 + mw * 32;
  const int wn0 = n0 + nw * 32;
#pragma unroll
  for (int t = 0; t < 4; ++t) {
    v8f acc = (t == 0) ? acc00 : (t == 1) ? acc01 : (t == 2) ? acc10 : acc11;
    const int mi = t >> 1, ni = t & 1;
    const int col = wn0 + ni * 16 + ln;
    const float bv = bias[col];
#pragma unroll
    for (int r = 0; r < 8; ++r) {
      const int row = wm0 + mi * 16 + r + 8 * hl;
      float v = acc[r] + bv;
      if (GELU_ACT) v = 0.5f * v * (1.f + erff(v * 0.70710678118654752f));
      if (RESID)    v += resid[(size_t)row * N + col];
      if (OUTMODE == 0) {
        outF[(size_t)row * N + col] = v;
      } else if (OUTMODE == 1) {
        outH[(size_t)row * N + col] = (_Float16)v;
      } else {
        // QKV scatter. col in [0,3D): sect 0=Q,1=K,2=V
        const int sect = col >> 10;
        const int dcol = col & (CD - 1);
        const int hh = dcol >> 6, dh = dcol & (CDH - 1);
        const int b = row >> 11, c = row & (CC - 1);
        const size_t bh = (size_t)b * CH + hh;
        if (sect == 0)      qo[(bh * CC + c) * CDH + dh] = (_Float16)(v * 0.125f); // fold 1/sqrt(64)
        else if (sect == 1) ko[(bh * CC + c) * CDH + dh] = (_Float16)v;
        else                vto[(bh * CDH + dh) * CC + c] = (_Float16)v;           // V transposed
      }
    }
  }
}

// ---------------------------------------------------------------------------
// Causal flash attention. Q: (b,h,C,dh) f16 (pre-scaled), K: (b,h,C,dh) f16,
// Vt: (b,h,dh,C) f16. Out attn: (b,C,h*dh) f16.
// One wave per 16-row Q tile; kv tiles of 32; online softmax.
// grid = (C/128, B*H), block = 256 (8 waves).
// ---------------------------------------------------------------------------
__global__ __launch_bounds__(256) void attn_kernel(const _Float16* __restrict__ Qb,
                                                   const _Float16* __restrict__ Kb,
                                                   const _Float16* __restrict__ Vt,
                                                   _Float16* __restrict__ attn) {
  __shared__ __align__(16) _Float16 Plds[8][16 * 32];
  const int lane = threadIdx.x & 31;
  const int wave = threadIdx.x >> 5;
  const int hl = lane >> 4, ln = lane & 15;
  const int bh = blockIdx.y;
  const int q0 = (blockIdx.x * 8 + wave) * 16;

  const _Float16* Q  = Qb + ((size_t)bh * CC + q0) * CDH;
  const _Float16* Kp = Kb + (size_t)bh * CC * CDH;
  const _Float16* Vp = Vt + (size_t)bh * CDH * CC;
  _Float16* Pl = &Plds[wave][0];

  const v16h qa0 = frag_ld(Q + 0,  CDH);   // dh 0..31
  const v16h qa1 = frag_ld(Q + 32, CDH);   // dh 32..63

  v8f o0 = {}, o1 = {}, o2 = {}, o3 = {};
  float mrow[8], lrow[8];
#pragma unroll
  for (int r = 0; r < 8; ++r) { mrow[r] = -1e30f; lrow[r] = 0.f; }

  for (int j0 = 0; j0 <= q0 + 15; j0 += 32) {
    // S tiles (16 q-rows x 32 kv-cols), K-dim = dh (64 -> two WMMAs each)
    v8f s0 = {}, s1 = {};
    s0 = WMMA_F16(qa0, frag_ld(Kp + (size_t)(j0)      * CDH + 0,  CDH), s0);
    s0 = WMMA_F16(qa1, frag_ld(Kp + (size_t)(j0)      * CDH + 32, CDH), s0);
    s1 = WMMA_F16(qa0, frag_ld(Kp + (size_t)(j0 + 16) * CDH + 0,  CDH), s1);
    s1 = WMMA_F16(qa1, frag_ld(Kp + (size_t)(j0 + 16) * CDH + 32, CDH), s1);

    // Causal mask + online softmax per row (row = q0 + r + 8*hl; cols over lanes 0..15)
#pragma unroll
    for (int r = 0; r < 8; ++r) {
      const int row = q0 + r + 8 * hl;
      float v0 = s0[r], v1 = s1[r];
      if (j0 + ln > row)      v0 = -1e30f;
      if (j0 + 16 + ln > row) v1 = -1e30f;
      float t = fmaxf(v0, v1);
#pragma unroll
      for (int off = 1; off < 16; off <<= 1) t = fmaxf(t, __shfl_xor(t, off, 32));
      const float mn = fmaxf(mrow[r], t);
      const float alpha = __expf(mrow[r] - mn);
      mrow[r] = mn;
      const float p0 = __expf(v0 - mn);
      const float p1 = __expf(v1 - mn);
      float ps = p0 + p1;
#pragma unroll
      for (int off = 1; off < 16; off <<= 1) ps += __shfl_xor(ps, off, 32);
      lrow[r] = lrow[r] * alpha + ps;
      o0[r] *= alpha; o1[r] *= alpha; o2[r] *= alpha; o3[r] *= alpha;
      Pl[(r + 8 * hl) * 32 + ln]      = (_Float16)p0;
      Pl[(r + 8 * hl) * 32 + 16 + ln] = (_Float16)p1;
    }

    // Stage P through wave-private LDS to re-read in A-fragment layout.
    asm volatile("s_wait_dscnt 0" ::: "memory");
    const v16h pa = frag_ld(Pl, 32);

    // O += P (16x32) @ V (32x64), four dh tiles of 16
    o0 = WMMA_F16(pa, frag_ld(Vp + (size_t)0  * CC + j0, CC), o0);
    o1 = WMMA_F16(pa, frag_ld(Vp + (size_t)16 * CC + j0, CC), o1);
    o2 = WMMA_F16(pa, frag_ld(Vp + (size_t)32 * CC + j0, CC), o2);
    o3 = WMMA_F16(pa, frag_ld(Vp + (size_t)48 * CC + j0, CC), o3);
  }

  const int b = bh >> 4, h = bh & 15;
  _Float16* out = attn + ((size_t)b * CC + q0) * CD + h * CDH;
#pragma unroll
  for (int r = 0; r < 8; ++r) {
    const float il = 1.0f / lrow[r];
    const size_t ro = (size_t)(r + 8 * hl) * CD;
    out[ro + 0  + ln] = (_Float16)(o0[r] * il);
    out[ro + 16 + ln] = (_Float16)(o1[r] * il);
    out[ro + 32 + ln] = (_Float16)(o2[r] * il);
    out[ro + 48 + ln] = (_Float16)(o3[r] * il);
  }
}

// ---------------------------------------------------------------------------
// Host-side orchestration
// ---------------------------------------------------------------------------
extern "C" void kernel_launch(void* const* d_in, const int* in_sizes, int n_in,
                              void* d_out, int out_size, void* d_ws, size_t ws_size,
                              hipStream_t stream) {
  (void)in_sizes; (void)n_in; (void)out_size; (void)ws_size;
  const float* x      = (const float*)d_in[0];
  const float* gamma1 = (const float*)d_in[1];
  const float* beta1  = (const float*)d_in[2];
  const float* W_qkv  = (const float*)d_in[3];
  const float* b_qkv  = (const float*)d_in[4];
  const float* W_o    = (const float*)d_in[5];
  const float* b_o    = (const float*)d_in[6];
  const float* gamma2 = (const float*)d_in[7];
  const float* beta2  = (const float*)d_in[8];
  const float* W1     = (const float*)d_in[9];
  const float* b1     = (const float*)d_in[10];
  const float* W2     = (const float*)d_in[11];
  const float* b2     = (const float*)d_in[12];
  float* out = (float*)d_out;

  // Workspace layout (linear allocator, 256B aligned)
  uint8_t* p = (uint8_t*)d_ws;
  auto take = [&](size_t n) { uint8_t* r = p; p += (n + 255) & ~(size_t)255; return r; };
  _Float16* Wqkv_t = (_Float16*)take((size_t)3 * CD * CD * 2);   //  6 MB (3072 x 1024)
  _Float16* Wo_t   = (_Float16*)take((size_t)CD * CD * 2);       //  2 MB (1024 x 1024)
  _Float16* W1_t   = (_Float16*)take((size_t)CHID * CD * 2);     //  8 MB (4096 x 1024)
  _Float16* W2_t   = (_Float16*)take((size_t)CD * CHID * 2);     //  8 MB (1024 x 4096)
  float*    x2     = (float*)take((size_t)CM * CD * 4);          // 32 MB
  _Float16* lnb    = (_Float16*)take((size_t)CM * CD * 2);       // 16 MB (reused ln1/ln2)
  uint8_t*  regA   = take((size_t)64 << 20);                     // 64 MB shared region
  _Float16* Qb   = (_Float16*)regA;                              // 16 MB
  _Float16* Kb   = (_Float16*)(regA + ((size_t)16 << 20));       // 16 MB
  _Float16* Vtb  = (_Float16*)(regA + ((size_t)32 << 20));       // 16 MB
  _Float16* attn = (_Float16*)(regA + ((size_t)48 << 20));       // 16 MB
  _Float16* hbuf = (_Float16*)regA;                              // 64 MB (after attention is dead)

  // 1) Weights -> f16, transposed to N-major
  wt_f16_kernel<<<(3 * CD * CD + 255) / 256, 256, 0, stream>>>(W_qkv, Wqkv_t, CD, 3 * CD);
  wt_f16_kernel<<<(CD * CD + 255) / 256,     256, 0, stream>>>(W_o,   Wo_t,   CD, CD);
  wt_f16_kernel<<<(CD * CHID + 255) / 256,   256, 0, stream>>>(W1,    W1_t,   CD, CHID);
  wt_f16_kernel<<<(CHID * CD + 255) / 256,   256, 0, stream>>>(W2,    W2_t,   CHID, CD);

  // 2) LN1
  layernorm_kernel<<<CM, 256, 0, stream>>>(x, gamma1, beta1, lnb);

  // 3) QKV GEMM (M=8192, N=3072, K=1024), scatter to Q/K/Vt
  gemm_wmma_kernel<2, false, false><<<dim3(3 * CD / 64, CM / 128), 256, 0, stream>>>(
      lnb, Wqkv_t, b_qkv, nullptr, nullptr, nullptr, Qb, Kb, Vtb, CM, 3 * CD, CD);

  // 4) Attention
  attn_kernel<<<dim3(CC / 128, CB * CH), 256, 0, stream>>>(Qb, Kb, Vtb, attn);

  // 5) O-proj + residual(x) -> x2 (f32)
  gemm_wmma_kernel<0, false, true><<<dim3(CD / 64, CM / 128), 256, 0, stream>>>(
      attn, Wo_t, b_o, x, x2, nullptr, nullptr, nullptr, nullptr, CM, CD, CD);

  // 6) LN2
  layernorm_kernel<<<CM, 256, 0, stream>>>(x2, gamma2, beta2, lnb);

  // 7) MLP1 + exact GELU -> hbuf (f16)
  gemm_wmma_kernel<1, true, false><<<dim3(CHID / 64, CM / 128), 256, 0, stream>>>(
      lnb, W1_t, b1, nullptr, nullptr, hbuf, nullptr, nullptr, nullptr, CM, CHID, CD);

  // 8) MLP2 + residual(x2) -> d_out (f32)
  gemm_wmma_kernel<0, false, true><<<dim3(CD / 64, CM / 128), 256, 0, stream>>>(
      hbuf, W2_t, b2, x2, out, nullptr, nullptr, nullptr, nullptr, CM, CD, CHID);
}
